// GATModel_8942121911043
// MI455X (gfx1250) — matrix operand order repaired and checked by reading, not compiled, not run on hardware
//
#include <hip/hip_runtime.h>
#include <math.h>

#define N_NODES 50000
#define N_EDGES 800000
#define HID     128
#define HEADS   8
#define DHEAD   16
#define NPF     32
#define NPAIRS  32768
#define HS_COLS 2336            // 288 + 4*512
#define BN_RS   0.9999950000374997f   // 1/sqrt(1 + 1e-5)

typedef __attribute__((ext_vector_type(16))) __bf16 v16bf;
typedef __attribute__((ext_vector_type(8)))  float  v8f;
typedef unsigned int u32x4 __attribute__((ext_vector_type(4)));
typedef int          i32x4 __attribute__((ext_vector_type(4)));
typedef int          i32x8 __attribute__((ext_vector_type(8)));

union Frag { v16bf v; unsigned u[8]; };

__device__ __forceinline__ unsigned short f2bf(float f) {
  unsigned u = __float_as_uint(f);
  u += 0x7FFFu + ((u >> 16) & 1u);      // round-to-nearest-even
  return (unsigned short)(u >> 16);
}
__device__ __forceinline__ unsigned fenc(float f) {   // order-preserving float->uint
  unsigned u = __float_as_uint(f);
  return (u & 0x80000000u) ? ~u : (u | 0x80000000u);
}
__device__ __forceinline__ float fdec(unsigned u) {
  return __uint_as_float((u & 0x80000000u) ? (u ^ 0x80000000u) : ~u);
}
__device__ __forceinline__ float eluf(float x) { return x > 0.f ? x : __expf(x) - 1.f; }

// ---------------------------------------------------------------------------
// One-time weight prep: Wt[n][k] = bf16(W[k][n])  (transpose + convert).
// Makes each GEMM B-tile a contiguous-row 2D bf16 tile -> TDM-friendly.
// ---------------------------------------------------------------------------
__global__ void k_w_to_bf16t(const float* __restrict__ W, int K, int N,
                             unsigned short* __restrict__ Wt) {
  int i = blockIdx.x * blockDim.x + threadIdx.x;
  if (i >= K * N) return;
  int n = i / K, k = i - n * K;
  Wt[(size_t)n * K + k] = f2bf(W[(size_t)k * N + n]);
}

// ---------------------------------------------------------------------------
// Tiled bf16 WMMA GEMM: C[:, col_off:col_off+N] (stride ldc) = act(A[M,K]) @ W[K,N] (+bias)
//  * A: f32, optional fused eval-BN (g*BN_RS, +b) + ReLU during LDS staging.
//  * B: pre-transposed bf16 weights Wt[N][K]; each 64x32 tile is DMA'd into LDS
//    by the Tensor Data Mover (tensor_load_to_lds) with HW pad for bank conflicts.
//  * 256 thr = 8 waves; 64x64 tile; 2x v_wmma_f32_16x16x32_bf16 per wave per K-step.
//  K must be a multiple of 32.
// ---------------------------------------------------------------------------
__global__ __launch_bounds__(256) void gemm_bf16_wmma(
    const float* __restrict__ A, int lda,
    const unsigned short* __restrict__ Bt, int ldbt,   // bf16 [N][K], row pitch = ldbt (=K)
    const float* __restrict__ g, const float* __restrict__ bvec, int use_bn,
    const float* __restrict__ bias,
    float* __restrict__ C, int ldc, int col_off,
    int M, int K, int N)
{
  __shared__ unsigned short As[64][40];   // [m][k] bf16, padded pitch (software)
  __shared__ unsigned short Bs[64][34];   // [n][k] bf16, 68B pitch via TDM pad_amount

  const int t        = threadIdx.x;
  const int blockRow = blockIdx.x * 64;
  const int colBlock = blockIdx.y * 64;

  const int w = t >> 5, lane = t & 31;
  const int wm = w & 3, wn = w >> 2;          // 4x2 wave grid over the 64x64 tile
  const int l15   = lane & 15;
  const int ahalf = (lane >> 4) * 8;          // A-fragment K sub-block select
  const int bhalf = (lane >> 4) * 16;         // B-fragment K sub-block select

  const int arow = t >> 2, acol = (t & 3) * 8;   // A staging: 4 thr/row x 8 floats

  v8f acc0 = {};
  v8f acc1 = {};

  for (int k0 = 0; k0 < K; k0 += 32) {
    __syncthreads();   // previous iteration's fragment reads complete

    // ---- wave 0: Tensor Data Mover DMA of B tile (64 rows x 32 bf16) -> LDS ----
    if (t < 32) {
      const unsigned long long gaddr =
          (unsigned long long)(const void*)(Bt + (size_t)colBlock * ldbt + k0);
      const unsigned ldsa = (unsigned)(unsigned long long)(const void*)&Bs[0][0];
      u32x4 g0;
      g0[0] = 1u;                                  // count=1, user descriptor
      g0[1] = ldsa;                                // lds_addr (bytes)
      g0[2] = (unsigned)gaddr;                     // global_addr[31:0]
      g0[3] = (unsigned)((gaddr >> 32) & 0x01FFFFFFu) | (2u << 30);  // addr[56:32] | type=2
      i32x8 g1;
      g1[0] = (int)((1u << 16) |                   // data_size = 2 bytes
                    (1u << 20) |                   // pad_enable
                    (3u << 22));                   // pad_interval: every 16 DWORDs (=64B row)
                                                   // pad_amount bits 31:25 = 0 -> 1 DWORD pad
      g1[1] = (int)(((unsigned)K & 0xFFFFu) << 16);               // tensor_dim0 lo16
      g1[2] = (int)((((unsigned)K >> 16) & 0xFFFFu) |
                    (((unsigned)N & 0xFFFFu) << 16));             // dim0 hi16 | tensor_dim1 lo16
      g1[3] = (int)((((unsigned)N >> 16) & 0xFFFFu) | (32u << 16)); // dim1 hi16 | tile_dim0=32
      g1[4] = (int)(64u);                                          // tile_dim1=64, tile_dim2=0
      g1[5] = (int)(unsigned)ldbt;                                 // tensor_dim0_stride lo32
      g1[6] = 0;                                                   // stride0 hi16 | stride1 lo16
      g1[7] = 0;
      i32x4 z4 = {0, 0, 0, 0};
#if __clang_major__ >= 23
      i32x8 z8 = {0, 0, 0, 0, 0, 0, 0, 0};
      __builtin_amdgcn_tensor_load_to_lds(g0, g1, z4, z4, z8, 0);
#else
      __builtin_amdgcn_tensor_load_to_lds(g0, g1, z4, z4, 0);
#endif
      __builtin_amdgcn_s_wait_tensorcnt(0);
    }

    // ---- stage A tile (64x32) with optional BN+ReLU, f32 -> bf16 ----
    {
      float va[8];
      const int gr = blockRow + arow;
      if (gr < M) {
        const float4* p = (const float4*)(A + (size_t)gr * lda + k0 + acol);
        float4 x0 = p[0], x1 = p[1];
        va[0]=x0.x; va[1]=x0.y; va[2]=x0.z; va[3]=x0.w;
        va[4]=x1.x; va[5]=x1.y; va[6]=x1.z; va[7]=x1.w;
      } else {
        #pragma unroll
        for (int j = 0; j < 8; ++j) va[j] = 0.f;
      }
      if (use_bn) {
        #pragma unroll
        for (int j = 0; j < 8; ++j) {
          const int kk = k0 + acol + j;
          float x = va[j] * (g[kk] * BN_RS) + bvec[kk];
          va[j] = x > 0.f ? x : 0.f;
        }
      }
      #pragma unroll
      for (int j = 0; j < 8; ++j) As[arow][acol + j] = f2bf(va[j]);
    }
    // prefetch next A tile into GL2 (global_prefetch_b8)
    if (k0 + 32 < K && blockRow + arow < M)
      __builtin_prefetch(A + (size_t)(blockRow + arow) * lda + k0 + 32 + acol, 0, 1);

    __syncthreads();   // A stores (DScnt) + TDM B tile visible to all waves

    // ---- per-lane fragment gather (ISA 7.12.2 layouts) + 2x WMMA ----
    Frag fa, fb0, fb1;
    const int am  = wm * 16 + l15;
    const int bn0 = wn * 32 + l15;
    #pragma unroll
    for (int v2 = 0; v2 < 8; ++v2) {
      const int ka = (v2 < 4 ? 2 * v2 : 16 + 2 * (v2 - 4)) + ahalf;  // A: K=(e&7)+(e>>3)*16+half*8
      const int kb = ka;                                              // B: K=(e&7)+(e>>3)*16+half*8
      fa.u[v2]  = *(const unsigned*)&As[am][ka];
      fb0.u[v2] = *(const unsigned*)&Bs[bn0][kb];
      fb1.u[v2] = *(const unsigned*)&Bs[bn0 + 16][kb];
    }
    acc0 = __builtin_amdgcn_wmma_f32_16x16x32_bf16(false, fa.v, false, fb0.v,
                                                   (short)0, acc0, false, false);
    acc1 = __builtin_amdgcn_wmma_f32_16x16x32_bf16(false, fa.v, false, fb1.v,
                                                   (short)0, acc1, false, false);
  }

  // ---- epilogue: C/D layout lane<16 -> M=r, lane>=16 -> M=r+8; N=lane&15 ----
  const int cn  = colBlock + wn * 32 + l15;
  const int cm0 = blockRow + wm * 16 + (lane >> 4) * 8;
  const float b0 = bias ? bias[cn]      : 0.f;
  const float b1 = bias ? bias[cn + 16] : 0.f;
  #pragma unroll
  for (int r = 0; r < 8; ++r) {
    const int m = cm0 + r;
    if (m < M) {
      float* cp = C + (size_t)m * ldc + col_off;
      cp[cn]      = acc0[r] + b0;
      cp[cn + 16] = acc1[r] + b1;
    }
  }
}

// ---------------------------------------------------------------------------
// Attention / scatter kernels
// ---------------------------------------------------------------------------
__global__ void k_fill_u32(unsigned* p, unsigned v, long long n) {
  long long i = (long long)blockIdx.x * blockDim.x + threadIdx.x;
  if (i < n) p[i] = v;
}

__global__ void k_el_er(const float* __restrict__ feat, const float* __restrict__ al,
                        const float* __restrict__ ar, float* __restrict__ el,
                        float* __restrict__ er, int H, int D) {
  int i = blockIdx.x * blockDim.x + threadIdx.x;
  if (i >= N_NODES * H) return;
  int n = i / H, h = i - n * H;
  const float* f = feat + (size_t)n * HID + h * D;
  float sl = 0.f, sr = 0.f;
  for (int d = 0; d < D; ++d) { float x = f[d]; sl += x * al[h * D + d]; sr += x * ar[h * D + d]; }
  el[i] = sl; er[i] = sr;
}

__global__ void k_edge_score_max(const int* __restrict__ src, const int* __restrict__ dst,
                                 const float* __restrict__ el, const float* __restrict__ er,
                                 float* __restrict__ ex, unsigned* __restrict__ menc, int H) {
  int i = blockIdx.x * blockDim.x + threadIdx.x;
  if (i >= N_EDGES * H) return;
  int e = i / H, h = i - e * H;
  float x = el[src[e] * H + h] + er[dst[e] * H + h];
  x = x > 0.f ? x : 0.2f * x;                       // leaky_relu(0.2)
  ex[i] = x;
  atomicMax(&menc[dst[e] * H + h], fenc(x));
}

__global__ void k_edge_expsum(const int* __restrict__ dst, float* __restrict__ ex,
                              const unsigned* __restrict__ menc, float* __restrict__ s, int H) {
  int i = blockIdx.x * blockDim.x + threadIdx.x;
  if (i >= N_EDGES * H) return;
  int e = i / H, h = i - e * H;
  float v = __expf(ex[i] - fdec(menc[dst[e] * H + h]));
  ex[i] = v;
  __hip_atomic_fetch_add(&s[dst[e] * H + h], v, __ATOMIC_RELAXED, __HIP_MEMORY_SCOPE_AGENT);
}

__global__ void k_aggregate(const int* __restrict__ src, const int* __restrict__ dst,
                            const float* __restrict__ ex, const float* __restrict__ s,
                            const float* __restrict__ feat, float* __restrict__ agg,
                            int H, int D) {
  int i = blockIdx.x * blockDim.x + threadIdx.x;
  if (i >= N_EDGES * H) return;
  int e = i / H, h = i - e * H;
  float a = ex[i] / (s[dst[e] * H + h] + 1e-9f);
  const float* f = feat + (size_t)src[e] * HID + h * D;
  float* o = agg + (size_t)dst[e] * HID + h * D;
  for (int d = 0; d < D; ++d)
    __hip_atomic_fetch_add(&o[d], f[d] * a, __ATOMIC_RELAXED, __HIP_MEMORY_SCOPE_AGENT);
}

// out = elu(bn(elu(agg + res + bias))) [+ extra_res]
__global__ void k_finalize(const float* __restrict__ agg, const float* __restrict__ res,
                           const float* __restrict__ bias, const float* __restrict__ g,
                           const float* __restrict__ b, const float* __restrict__ extra_res,
                           float* __restrict__ out) {
  int i = blockIdx.x * blockDim.x + threadIdx.x;
  if (i >= N_NODES * HID) return;
  int c = i & (HID - 1);
  float v = agg[i] + res[i] + bias[c];
  v = eluf(v);
  v = v * (g[c] * BN_RS) + b[c];
  v = eluf(v);
  if (extra_res) v += extra_res[i];
  out[i] = v;
}

__global__ void k_pair_gather(const int* __restrict__ vp, const float* __restrict__ gf,
                              const float* __restrict__ pf, float* __restrict__ hs) {
  int i = blockIdx.x * blockDim.x + threadIdx.x;
  if (i >= NPAIRS * 288) return;
  int p = i / 288, c = i - p * 288;
  float v;
  if (c < 128)      v = gf[(size_t)vp[2 * p]     * HID + c];
  else if (c < 256) v = gf[(size_t)vp[2 * p + 1] * HID + (c - 128)];
  else              v = pf[(size_t)p * NPF + (c - 256)];
  hs[(size_t)p * HS_COLS + c] = v;
}

// final dense block: N=1 -> wave-per-row GEMV with shuffle reduction
__global__ __launch_bounds__(256) void k_final_pred(
    const float* __restrict__ hs, const float* __restrict__ g, const float* __restrict__ b,
    const float* __restrict__ W, const float* __restrict__ bias, float* __restrict__ out) {
  int p = blockIdx.x * 8 + (threadIdx.x >> 5);
  int lane = threadIdx.x & 31;
  const float* row = hs + (size_t)p * HS_COLS;
  float acc = 0.f;
  for (int k = lane; k < HS_COLS; k += 32) {
    float x = row[k] * (g[k] * BN_RS) + b[k];
    x = x > 0.f ? x : 0.f;
    acc += x * W[k];
  }
  #pragma unroll
  for (int o = 16; o; o >>= 1) acc += __shfl_xor(acc, o, 32);
  if (lane == 0) out[p] = acc + bias[0];
}

// ---------------------------------------------------------------------------
extern "C" void kernel_launch(void* const* d_in, const int* in_sizes, int n_in,
                              void* d_out, int out_size, void* d_ws, size_t ws_size,
                              hipStream_t stream) {
  (void)in_sizes; (void)n_in; (void)out_size; (void)ws_size;
  // setup_inputs() order: node_features(ignored: identity gather), edge_src, edge_dst,
  // vertex_pairs, pairwise_features, then params flattened in insertion order.
  const int*   esrc  = (const int*)  d_in[1];
  const int*   edst  = (const int*)  d_in[2];
  const int*   vp    = (const int*)  d_in[3];
  const float* pf    = (const float*)d_in[4];
  const float* emb   = (const float*)d_in[5];
  const float* W1    = (const float*)d_in[6];
  const float* al1   = (const float*)d_in[7];
  const float* ar1   = (const float*)d_in[8];
  const float* bias1 = (const float*)d_in[9];
  const float* W2    = (const float*)d_in[10];
  const float* al2   = (const float*)d_in[11];
  const float* ar2   = (const float*)d_in[12];
  const float* bias2 = (const float*)d_in[13];
  const float* bn1g  = (const float*)d_in[14];
  const float* bn1b  = (const float*)d_in[15];
  const float* bn2g  = (const float*)d_in[16];
  const float* bn2b  = (const float*)d_in[17];
  const float *dg[5], *db[5], *dW[5], *dbias[5];
  for (int i = 0; i < 5; ++i) {
    dg[i]    = (const float*)d_in[18 + 4 * i];
    db[i]    = (const float*)d_in[19 + 4 * i];
    dW[i]    = (const float*)d_in[20 + 4 * i];
    dbias[i] = (const float*)d_in[21 + 4 * i];
  }

  // workspace layout (256B-aligned chunks)
  char* base = (char*)d_ws;
  size_t off = 0;
  auto allocb = [&](size_t bytes) {
    void* p = base + off;
    off += (bytes + 255) & ~(size_t)255;
    return p;
  };
  const size_t NH = (size_t)N_NODES * HID;
  const int din[4] = {288, 800, 1312, 1824};   // all multiples of 32

  float*    feat = (float*)   allocb(NH * 4);
  float*    agg  = (float*)   allocb(NH * 4);
  float*    h1   = (float*)   allocb(NH * 4);
  float*    gf   = (float*)   allocb(NH * 4);
  float*    el   = (float*)   allocb((size_t)N_NODES * HEADS * 4);
  float*    er   = (float*)   allocb((size_t)N_NODES * HEADS * 4);
  unsigned* menc = (unsigned*)allocb((size_t)N_NODES * HEADS * 4);
  float*    sbuf = (float*)   allocb((size_t)N_NODES * HEADS * 4);
  float*    ex   = (float*)   allocb((size_t)N_EDGES * HEADS * 4);
  float*    hs   = (float*)   allocb((size_t)NPAIRS * HS_COLS * 4);
  unsigned short* Wt1 = (unsigned short*)allocb((size_t)HID * HID * 2);
  unsigned short* Wt2 = (unsigned short*)allocb((size_t)HID * HID * 2);
  unsigned short* Wtd[4];
  for (int i = 0; i < 4; ++i) Wtd[i] = (unsigned short*)allocb((size_t)512 * din[i] * 2);

  const dim3 blk(256);
  auto cdiv = [](long long a, long long b) { return (unsigned)((a + b - 1) / b); };

  // ---- one-time weight transpose + bf16 convert (B matrices for all GEMMs) ----
  k_w_to_bf16t<<<cdiv(HID * HID, 256), blk, 0, stream>>>(W1, HID, HID, Wt1);
  k_w_to_bf16t<<<cdiv(HID * HID, 256), blk, 0, stream>>>(W2, HID, HID, Wt2);
  for (int i = 0; i < 4; ++i)
    k_w_to_bf16t<<<cdiv((long long)din[i] * 512, 256), blk, 0, stream>>>(dW[i], din[i], 512, Wtd[i]);

  // ============================ GAT layer 1 (H=8, D=16) ============================
  gemm_bf16_wmma<<<dim3(cdiv(N_NODES, 64), HID / 64), blk, 0, stream>>>(
      emb, HID, Wt1, HID, nullptr, nullptr, 0, nullptr, feat, HID, 0, N_NODES, HID, HID);
  k_el_er<<<cdiv((long long)N_NODES * HEADS, 256), blk, 0, stream>>>(feat, al1, ar1, el, er, HEADS, DHEAD);
  k_fill_u32<<<cdiv((long long)N_NODES * HEADS, 256), blk, 0, stream>>>(menc, 0u, (long long)N_NODES * HEADS);
  k_fill_u32<<<cdiv((long long)N_NODES * HEADS, 256), blk, 0, stream>>>((unsigned*)sbuf, 0u, (long long)N_NODES * HEADS);
  k_fill_u32<<<cdiv((long long)NH, 256), blk, 0, stream>>>((unsigned*)agg, 0u, (long long)NH);
  k_edge_score_max<<<cdiv((long long)N_EDGES * HEADS, 256), blk, 0, stream>>>(esrc, edst, el, er, ex, menc, HEADS);
  k_edge_expsum<<<cdiv((long long)N_EDGES * HEADS, 256), blk, 0, stream>>>(edst, ex, menc, sbuf, HEADS);
  k_aggregate<<<cdiv((long long)N_EDGES * HEADS, 256), blk, 0, stream>>>(esrc, edst, ex, sbuf, feat, agg, HEADS, DHEAD);
  k_finalize<<<cdiv((long long)NH, 256), blk, 0, stream>>>(agg, emb, bias1, bn1g, bn1b, nullptr, h1);

  // ============================ GAT layer 2 (H=1, D=128) ===========================
  gemm_bf16_wmma<<<dim3(cdiv(N_NODES, 64), HID / 64), blk, 0, stream>>>(
      h1, HID, Wt2, HID, nullptr, nullptr, 0, nullptr, feat, HID, 0, N_NODES, HID, HID);
  k_el_er<<<cdiv((long long)N_NODES, 256), blk, 0, stream>>>(feat, al2, ar2, el, er, 1, HID);
  k_fill_u32<<<cdiv((long long)N_NODES, 256), blk, 0, stream>>>(menc, 0u, (long long)N_NODES);
  k_fill_u32<<<cdiv((long long)N_NODES, 256), blk, 0, stream>>>((unsigned*)sbuf, 0u, (long long)N_NODES);
  k_fill_u32<<<cdiv((long long)NH, 256), blk, 0, stream>>>((unsigned*)agg, 0u, (long long)NH);
  k_edge_score_max<<<cdiv((long long)N_EDGES, 256), blk, 0, stream>>>(esrc, edst, el, er, ex, menc, 1);
  k_edge_expsum<<<cdiv((long long)N_EDGES, 256), blk, 0, stream>>>(edst, ex, menc, sbuf, 1);
  k_aggregate<<<cdiv((long long)N_EDGES, 256), blk, 0, stream>>>(esrc, edst, ex, sbuf, feat, agg, 1, HID);
  k_finalize<<<cdiv((long long)NH, 256), blk, 0, stream>>>(agg, h1, bias2, bn2g, bn2b, h1, gf);

  // ============================ DenseNet MLP head ==================================
  k_pair_gather<<<cdiv((long long)NPAIRS * 288, 256), blk, 0, stream>>>(vp, gf, pf, hs);
  for (int i = 0; i < 4; ++i) {
    gemm_bf16_wmma<<<dim3(NPAIRS / 64, 512 / 64), blk, 0, stream>>>(
        hs, HS_COLS, Wtd[i], din[i], dg[i], db[i], 1, dbias[i],
        hs, HS_COLS, din[i], NPAIRS, din[i], 512);
  }
  k_final_pred<<<NPAIRS / 8, blk, 0, stream>>>(hs, dg[4], db[4], dW[4], dbias[4], (float*)d_out);
}
